// NeuralNetwork_32504312496328
// MI455X (gfx1250) — matrix operand order repaired and checked by reading, not compiled, not run on hardware
//
#include <hip/hip_runtime.h>
#include <hip/hip_bf16.h>
#include <math.h>

// ---------------- CDNA5 WMMA types ----------------
typedef __attribute__((ext_vector_type(16))) __bf16 v16bf;
typedef __attribute__((ext_vector_type(8)))  __bf16 v8bf;
typedef __attribute__((ext_vector_type(8)))  float  v8f;

static __device__ __forceinline__ __bf16 to_bf16(float f) { return (__bf16)f; }
static __device__ __forceinline__ float sigmoidf_(float x) { return 1.0f / (1.0f + expf(-x)); }
static __device__ __forceinline__ void pack4(v16bf& a, int o, float4 r) {
    a[o + 0] = to_bf16(r.x); a[o + 1] = to_bf16(r.y);
    a[o + 2] = to_bf16(r.z); a[o + 3] = to_bf16(r.w);
}

#define EPSV 1e-5f

// =====================================================================
// Kernel 1: conv1 (k=s=4, 3->16ch) + BN + ReLU + 2x2 maxpool, fused.
// stride==kernel => GEMM over non-overlapping 4x4x3 patches:
//   A = 16 patches x 48 (pad 64) bf16, B = 48x16 weights, D = 16x16 f32.
// Each wave owns a 4x4 block of pooled outputs x 16 channels; 4 WMMA
// sub-tiles (one per pool position) are BN+ReLU'd and maxed.
// A-gather uses 16B-aligned float4 (global_load_b128): the ISA A layout
//   K = h + (h>=8?8:0) + (hiHalf?8:0)   maps to
//   ch = {0,1,2}, dy = hiHalf*2 + (h&4?1:0), dx = h&3  -> contiguous rows.
// Output NHWC [64][96][96][16]: lanes 0-15 store contiguous 64B.
// =====================================================================
__global__ __launch_bounds__(256) void conv1_bn_pool_kernel(
    const float* __restrict__ x, const float* __restrict__ w,
    const float* __restrict__ cb, const float* __restrict__ gg,
    const float* __restrict__ bt, const float* __restrict__ mn,
    const float* __restrict__ vr, float* __restrict__ pool1)
{
    int wave = (blockIdx.x * blockDim.x + threadIdx.x) >> 5;
    int lane = threadIdx.x & 31;
    int hiHalf = lane >> 4;
    int n = lane & 15;                 // output channel (B/C column)

    int b   = wave / 576;              // 24*24 tiles per sample
    int t   = wave % 576;
    int pi0 = (t / 24) * 4;
    int pj0 = (t % 24) * 4;

    // --- B matrices (weights), K x N = 64x16 in two K=32 halves.
    // Layout: K = h + (h>=8?8:0) + hiHalf*8 (per half), matching A.
    v16bf wb0, wb1;
    #pragma unroll
    for (int h = 0; h < 16; ++h) {
        int k0 = h + (h >= 8 ? 8 : 0) + (hiHalf ? 8 : 0);
        int k1 = k0 + 32;
        wb0[h] = to_bf16(w[n * 48 + k0]);                        // K<48 always
        wb1[h] = (k1 < 48) ? to_bf16(w[n * 48 + k1]) : to_bf16(0.0f);
    }

    // Per-channel BN affine: sc*h + sh (h includes conv bias)
    float sc = gg[n] * rsqrtf(vr[n] + EPSV);
    float sh = bt[n] - mn[n] * sc;
    float cbias = cb[n];

    const float* xb = x + (size_t)b * 3 * 768 * 768;

    // prefetch input tile (HBM-bound stage) -> global_prefetch_b8
    #pragma unroll
    for (int ch = 0; ch < 3; ++ch)
        __builtin_prefetch(xb + ((size_t)ch * 768 + (size_t)pi0 * 8) * 768 + (size_t)pj0 * 8, 0, 3);

    int mrow = lane & 15;              // A row = patch index within tile
    int pi = pi0 + (mrow >> 2);
    int pj = pj0 + (mrow & 3);

    const size_t CH = (size_t)768 * 768;   // channel plane stride (floats)
    int dyB = hiHalf * 2;                  // this lane-half covers rows dyB, dyB+1

    v8f best;
    #pragma unroll
    for (int s = 0; s < 4; ++s) {
        int ci = 2 * pi + (s >> 1);
        int cj = 2 * pj + (s & 1);
        const float* base = xb + (size_t)(ci * 4 + dyB) * 768 + cj * 4;

        // 6x global_load_b128 per lane: {ch0,ch1,ch2} x {row dyB, row dyB+1}
        float4 r00 = *(const float4*)(base);
        float4 r01 = *(const float4*)(base + 768);
        float4 r10 = *(const float4*)(base + CH);
        float4 r11 = *(const float4*)(base + CH + 768);
        float4 r20 = *(const float4*)(base + 2 * CH);
        float4 r21 = *(const float4*)(base + 2 * CH + 768);

        v16bf a0, a1 = {};
        pack4(a0, 0, r00); pack4(a0, 4, r01);    // halves 0..7  : ch0 (K 0..15)
        pack4(a0, 8, r10); pack4(a0, 12, r11);   // halves 8..15 : ch1 (K 16..31)
        pack4(a1, 0, r20); pack4(a1, 4, r21);    // halves 0..7  : ch2 (K 32..47)
                                                  // halves 8..15 : zero (K 48..63 pad)
        v8f acc = {};
        acc = __builtin_amdgcn_wmma_f32_16x16x32_bf16(false, a0, false, wb0, (short)0, acc, false, false);
        acc = __builtin_amdgcn_wmma_f32_16x16x32_bf16(false, a1, false, wb1, (short)0, acc, false, false);

        #pragma unroll
        for (int v = 0; v < 8; ++v) {
            float r = fmaxf(sc * (acc[v] + cbias) + sh, 0.0f);   // BN + ReLU
            best[v] = (s == 0) ? r : fmaxf(best[v], r);          // maxpool
        }
    }

    // store NHWC: D layout M = v + hiHalf*8, N = lane&15
    #pragma unroll
    for (int v = 0; v < 8; ++v) {
        int m  = v + (hiHalf ? 8 : 0);
        int qi = pi0 + (m >> 2), qj = pj0 + (m & 3);
        pool1[(((size_t)b * 96 + qi) * 96 + qj) * 16 + n] = best[v];
    }
}

// =====================================================================
// Kernel 2: conv2 (k=s=3, 16->1ch) + BN + ReLU + 2x2 maxpool -> c[64][256]
// pool1 is NHWC, 64B-aligned per pixel -> 4x float4 loads per tap.
// =====================================================================
__global__ __launch_bounds__(256) void conv2_bn_pool_kernel(
    const float* __restrict__ pool1, const float* __restrict__ w2,
    const float* __restrict__ cb2, const float* __restrict__ g2,
    const float* __restrict__ bt2, const float* __restrict__ m2,
    const float* __restrict__ v2, float* __restrict__ cfeat)
{
    __shared__ float wsm[144];                 // transposed: [(dy*3+dx)][ch]
    int tid = threadIdx.x;
    if (tid < 144) wsm[(tid % 9) * 16 + tid / 9] = w2[tid];   // w2: [ch][dy][dx]
    __syncthreads();

    int idx = blockIdx.x * 256 + tid;          // 64*256 pooled outputs
    int b = idx >> 8;
    int p = idx & 255;
    int pi = p >> 4, pj = p & 15;

    float sc = g2[0] * rsqrtf(v2[0] + EPSV);
    float sh = bt2[0] - m2[0] * sc;
    float bias = cb2[0];

    float best = -INFINITY;
    #pragma unroll
    for (int s = 0; s < 4; ++s) {
        int ci = 2 * pi + (s >> 1);
        int cj = 2 * pj + (s & 1);
        float acc = 0.0f;
        #pragma unroll
        for (int dy = 0; dy < 3; ++dy)
            #pragma unroll
            for (int dx = 0; dx < 3; ++dx) {
                const float4* s4 = (const float4*)(pool1 +
                    (((size_t)b * 96 + (ci * 3 + dy)) * 96 + (cj * 3 + dx)) * 16);
                const float* wv = wsm + (dy * 3 + dx) * 16;
                float4 p0 = s4[0], p1 = s4[1], p2 = s4[2], p3 = s4[3];
                acc += p0.x * wv[0]  + p0.y * wv[1]  + p0.z * wv[2]  + p0.w * wv[3];
                acc += p1.x * wv[4]  + p1.y * wv[5]  + p1.z * wv[6]  + p1.w * wv[7];
                acc += p2.x * wv[8]  + p2.y * wv[9]  + p2.z * wv[10] + p2.w * wv[11];
                acc += p3.x * wv[12] + p3.y * wv[13] + p3.z * wv[14] + p3.w * wv[15];
            }
        best = fmaxf(best, fmaxf(sc * (acc + bias) + sh, 0.0f));
    }
    cfeat[idx] = best;                         // matches reshape(b, 256)
}

// =====================================================================
// Kernel 3: router MLP 256->32->32->3, argmax (softmax is monotone).
// =====================================================================
__global__ __launch_bounds__(32) void route_kernel(
    const float* __restrict__ cfeat,
    const float* __restrict__ w1, const float* __restrict__ b1,
    const float* __restrict__ w2, const float* __restrict__ b2,
    const float* __restrict__ w3, const float* __restrict__ b3,
    int* __restrict__ route)
{
    __shared__ float l1[32], l2[32], lg[3];
    int b = blockIdx.x, t = threadIdx.x;
    const float* c = cfeat + (size_t)b * 256;

    float acc = b1[t];
    for (int k = 0; k < 256; ++k) acc += c[k] * w1[t * 256 + k];
    l1[t] = fmaxf(acc, 0.0f);
    __syncthreads();

    acc = b2[t];
    #pragma unroll
    for (int k = 0; k < 32; ++k) acc += l1[k] * w2[t * 32 + k];
    l2[t] = fmaxf(acc, 0.0f);
    __syncthreads();

    if (t < 3) {
        acc = b3[t];
        #pragma unroll
        for (int k = 0; k < 32; ++k) acc += l2[k] * w3[t * 32 + k];
        lg[t] = acc;
    }
    __syncthreads();
    if (t == 0) {
        int r = 0; float m = lg[0];
        if (lg[1] > m) { m = lg[1]; r = 1; }
        if (lg[2] > m) { r = 2; }
        route[b] = r;
    }
}

// =====================================================================
// Kernel 4: LSTM experts. 1 block (16 waves) per expert, 256 sequential
// steps. Per step: gates[64x256] = h[64x64] @ w_hh^T as bf16 WMMA
// (4 M-tiles x 2 K-steps per wave = 8 v_wmma; 128/step/block).
// w_hh resident in registers for all 256 steps; h bf16 / c f32 in LDS.
// A-gather = 2x ds_load_b128 per K-half; x(t+1) prefetch folded into the
// update phase => 2 barriers/step on the serial critical path.
// =====================================================================
__global__ __launch_bounds__(512) void lstm_kernel(
    const float* __restrict__ cfeat,
    const float* __restrict__ w_ih, const float* __restrict__ w_hh,
    const float* __restrict__ b_ih, const float* __restrict__ b_hh,
    float* __restrict__ outs)
{
    __shared__ __bf16 hA[64 * 64];     // h state [b][j] bf16 (A source)
    __shared__ float  cst[64 * 64];    // c state f32
    __shared__ float  gt[64 * 256];    // gates f32 [b][4H]
    __shared__ float  wih[256];
    __shared__ float  biasS[256];
    __shared__ float  xcur[64];

    int e   = blockIdx.x;
    int tid = threadIdx.x;
    int wv  = tid >> 5;                // 0..15 -> N-tile (16 gates)
    int lane = tid & 31;
    int hiHalf = lane >> 4;
    int gcol = wv * 16 + (lane & 15);  // gate column owned in B/C

    // resident B registers: w_hh[gcol][0..63] in two K=32 halves
    v16bf wb0, wb1;
    {
        const float* whh = w_hh + (size_t)e * 256 * 64 + (size_t)gcol * 64;
        #pragma unroll
        for (int h = 0; h < 16; ++h) {
            int k0 = h + (h >= 8 ? 8 : 0) + (hiHalf ? 8 : 0);
            wb0[h] = to_bf16(whh[k0]);
            wb1[h] = to_bf16(whh[k0 + 32]);
        }
    }
    if (tid < 256) {
        wih[tid]   = w_ih[e * 256 + tid];
        biasS[tid] = b_ih[e * 256 + tid] + b_hh[e * 256 + tid];
    }
    if (tid < 64) xcur[tid] = cfeat[(size_t)tid * 256];     // x for t=0
    for (int i = tid; i < 64 * 64; i += 512) { hA[i] = to_bf16(0.0f); cst[i] = 0.0f; }
    __syncthreads();

    for (int t = 0; t < 256; ++t) {
        // ---- gates = h @ w_hh^T (+ x_t*w_ih + bias folded on store) ----
        #pragma unroll
        for (int mt = 0; mt < 4; ++mt) {
            int brow = mt * 16 + (lane & 15);
            const __bf16* hrow = hA + brow * 64;
            // A layout: K = h + (h>=8?8:0) + hiHalf*8 -> two 16B ds reads/half
            v8bf lo0 = *(const v8bf*)(hrow + hiHalf * 8);
            v8bf hi0 = *(const v8bf*)(hrow + 16 + hiHalf * 8);
            v8bf lo1 = *(const v8bf*)(hrow + 32 + hiHalf * 8);
            v8bf hi1 = *(const v8bf*)(hrow + 48 + hiHalf * 8);
            v16bf a0 = __builtin_shufflevector(lo0, hi0, 0,1,2,3,4,5,6,7,8,9,10,11,12,13,14,15);
            v16bf a1 = __builtin_shufflevector(lo1, hi1, 0,1,2,3,4,5,6,7,8,9,10,11,12,13,14,15);

            v8f acc = {};
            acc = __builtin_amdgcn_wmma_f32_16x16x32_bf16(false, a0, false, wb0, (short)0, acc, false, false);
            acc = __builtin_amdgcn_wmma_f32_16x16x32_bf16(false, a1, false, wb1, (short)0, acc, false, false);
            #pragma unroll
            for (int v = 0; v < 8; ++v) {
                int bs = mt * 16 + v + (hiHalf ? 8 : 0);    // C/D: M = v + hiHalf*8
                gt[bs * 256 + gcol] = acc[v] + xcur[bs] * wih[gcol] + biasS[gcol];
            }
        }
        __syncthreads();

        // ---- cell update (4096 (b,j) pairs) + x prefetch for t+1 ----
        for (int p = tid; p < 64 * 64; p += 512) {
            int bs = p >> 6, j = p & 63;
            float ig = gt[bs * 256 + j];
            float fg = gt[bs * 256 + 64 + j];
            float gv = gt[bs * 256 + 128 + j];
            float og = gt[bs * 256 + 192 + j];
            float cc = sigmoidf_(fg) * cst[p] + sigmoidf_(ig) * tanhf(gv);
            cst[p] = cc;
            float hh = sigmoidf_(og) * tanhf(cc);
            hA[p] = to_bf16(hh);
            if (j == 63) outs[((size_t)e * 64 + bs) * 256 + t] = hh;   // h[:, -1]
        }
        if (tid < 64 && t + 1 < 256) xcur[tid] = cfeat[(size_t)tid * 256 + t + 1];
        __syncthreads();
    }
}

// =====================================================================
// Kernel 5: gather routed expert + linear head -> out[64][3]
// =====================================================================
__global__ __launch_bounds__(192) void head_kernel(
    const float* __restrict__ outs, const int* __restrict__ route,
    const float* __restrict__ sw, const float* __restrict__ sb,
    float* __restrict__ out)
{
    int tid = threadIdx.x;
    int b = tid / 3, k = tid % 3;
    int r = route[b];
    const float* o = outs + ((size_t)r * 64 + b) * 256;
    float acc = sb[k];
    for (int t = 0; t < 256; ++t) acc += o[t] * sw[k * 256 + t];
    out[b * 3 + k] = acc;
}

// =====================================================================
extern "C" void kernel_launch(void* const* d_in, const int* in_sizes, int n_in,
                              void* d_out, int out_size, void* d_ws, size_t ws_size,
                              hipStream_t stream) {
    (void)in_sizes; (void)n_in; (void)out_size; (void)ws_size;

    const float* x       = (const float*)d_in[0];
    const float* conv1_w = (const float*)d_in[1];
    const float* conv1_b = (const float*)d_in[2];
    const float* bn1_g   = (const float*)d_in[3];
    const float* bn1_b   = (const float*)d_in[4];
    const float* bn1_m   = (const float*)d_in[5];
    const float* bn1_v   = (const float*)d_in[6];
    const float* conv2_w = (const float*)d_in[7];
    const float* conv2_b = (const float*)d_in[8];
    const float* bn2_g   = (const float*)d_in[9];
    const float* bn2_b   = (const float*)d_in[10];
    const float* bn2_m   = (const float*)d_in[11];
    const float* bn2_v   = (const float*)d_in[12];
    const float* pre_w1  = (const float*)d_in[13];
    const float* pre_b1  = (const float*)d_in[14];
    const float* pre_w2  = (const float*)d_in[15];
    const float* pre_b2  = (const float*)d_in[16];
    const float* pre_w3  = (const float*)d_in[17];
    const float* pre_b3  = (const float*)d_in[18];
    const float* lstm_w_ih = (const float*)d_in[19];
    const float* lstm_w_hh = (const float*)d_in[20];
    const float* lstm_b_ih = (const float*)d_in[21];
    const float* lstm_b_hh = (const float*)d_in[22];
    const float* stack_w   = (const float*)d_in[23];
    const float* stack_b   = (const float*)d_in[24];

    // workspace layout (~38 MB)
    char* ws = (char*)d_ws;
    size_t off = 0;
    float* pool1 = (float*)(ws + off); off += (size_t)64 * 96 * 96 * 16 * sizeof(float);
    float* cfeat = (float*)(ws + off); off += (size_t)64 * 256 * sizeof(float);
    int*   route = (int*)(ws + off);   off += 256;
    float* outs  = (float*)(ws + off); off += (size_t)3 * 64 * 256 * sizeof(float);

    // conv1: 64 samples * 24*24 tiles = 36864 waves, 8 waves/block
    conv1_bn_pool_kernel<<<4608, 256, 0, stream>>>(
        x, conv1_w, conv1_b, bn1_g, bn1_b, bn1_m, bn1_v, pool1);

    conv2_bn_pool_kernel<<<64, 256, 0, stream>>>(
        pool1, conv2_w, conv2_b, bn2_g, bn2_b, bn2_m, bn2_v, cfeat);

    route_kernel<<<64, 32, 0, stream>>>(
        cfeat, pre_w1, pre_b1, pre_w2, pre_b2, pre_w3, pre_b3, route);

    lstm_kernel<<<3, 512, 0, stream>>>(
        cfeat, lstm_w_ih, lstm_w_hh, lstm_b_ih, lstm_b_hh, outs);

    head_kernel<<<1, 192, 0, stream>>>(
        outs, route, stack_w, stack_b, (float*)d_out);
}
